// Dominant_9328668967790
// MI455X (gfx1250) — matrix operand order repaired and checked by reading, not compiled, not run on hardware
//
#include <hip/hip_runtime.h>
#include <hip/hip_bf16.h>

// ---------------------------------------------------------------------------
// Dominant graph-VAE forward, MI455X (gfx1250, wave32, WMMA).
// All matmuls via v_wmma_f32_16x16x32_f16 (f16 in-register convert, f32 acc).
// HBM-bound: adj streamed twice (512MB, NT loads), A_hat stored once (256MB,
// NT stores). Register-prefetch software pipeline overlaps global loads with
// the WMMA compute phase.
// ---------------------------------------------------------------------------

typedef float    f32x4 __attribute__((ext_vector_type(4)));
typedef _Float16 half4 __attribute__((ext_vector_type(4)));
typedef _Float16 half8 __attribute__((ext_vector_type(8)));
typedef _Float16 v16h  __attribute__((ext_vector_type(16)));
typedef float    v8f   __attribute__((ext_vector_type(8)));

#define NROWS 128   // rows of C per block (8 waves x 16 rows)
#define KS    32    // K-step (WMMA f16 K)

// C[M, NW-slab] = act( cscale * (A@B) + bias )
//   A: fp32 [M,K] row-major (lda%4==0, K%32==0, M%128==0), converted to f16
//      with pre-scale `ascale` (undone by cscale in f32 epilogue).
//   B: fp32, element (k,n) at B[k*bk + n*bn]; bn==1 (row-major) or bk==1
//      (transposed, for s@s^T). Staged into LDS transposed [n][k].
//   ANT: non-temporal A loads (streaming adj). CNT: non-temporal C stores.
//   act: 0=none 1=relu 2=sigmoid
template<int NW, bool ANT, bool CNT>
__global__ __launch_bounds__(256)
void gemm_wmma_f16(const float* __restrict__ A, int lda,
                   const float* __restrict__ B, int bk, int bn,
                   const float* __restrict__ bias,
                   float* __restrict__ C, int ldc,
                   int K, int act, float ascale, float cscale)
{
    constexpr int NTILE = NW / 16;         // 16-wide N tiles per wave
    constexpr int NB    = NW / 32;         // B float4 slots per thread
    __shared__ _Float16 As[NROWS * 48];    // 128x32 f16, stride 48 (16B frags)
    __shared__ _Float16 Bs[NW * 48];       // transposed [n][k], stride 48

    const int t     = threadIdx.x;
    const int lane  = t & 31;
    const int wave  = t >> 5;
    const int m0    = blockIdx.x * NROWS;
    const int n0    = blockIdx.y * NW;
    const int lhalf = lane >> 4;           // 0: lanes 0-15, 1: lanes 16-31
    const int lmod  = lane & 15;
    const bool brow = (bn == 1);

    // ---- per-thread staging slots (computed once) ----
    const float* aPtr[4];
    int aLds[4];
    #pragma unroll
    for (int i = 0; i < 4; ++i) {
        const int idx = t + 256 * i;       // 0..1023 float4 slots of 128x32
        const int row = idx >> 3;
        const int c4  = idx & 7;
        aPtr[i] = A + (size_t)(m0 + row) * lda + c4 * 4;
        aLds[i] = row * 48 + c4 * 4;
    }
    const float* bPtr[NB];
    int bLds[NB];                          // base LDS offset per slot
    long bStep;                            // pointer advance per K-step
    if (brow) {
        bStep = (long)KS * bk;
        #pragma unroll
        for (int i = 0; i < NB; ++i) {
            const int idx = t + 256 * i;   // 8*NW float4 slots of 32xNW
            const int n4  = idx % (NW / 4);
            const int kk  = idx / (NW / 4);
            bPtr[i] = B + (size_t)kk * bk + n0 + n4 * 4;
            bLds[i] = (n4 * 4) * 48 + kk;  // +q*48 for q=0..3
        }
    } else {                               // bk==1: (k,n) at n*bn + k
        bStep = (long)KS;
        #pragma unroll
        for (int i = 0; i < NB; ++i) {
            const int idx = t + 256 * i;
            const int k4  = idx & 7;
            const int nn  = idx >> 3;
            bPtr[i] = B + (size_t)(n0 + nn) * bn + k4 * 4;
            bLds[i] = nn * 48 + k4 * 4;    // contiguous half4
        }
    }

    // ---- LDS fragment addresses (computed once) ----
    const int arow   = wave * 16 + lmod;
    const int aFrag0 = arow * 48 + lhalf * 8;        // + {0, 16}
    const int bFrag0 = lmod * 48 + lhalf * 8;        // + nt*16*48, + {0, 16}

    v8f acc[NTILE];
    #pragma unroll
    for (int i = 0; i < NTILE; ++i)
        #pragma unroll
        for (int r = 0; r < 8; ++r) acc[i][r] = 0.0f;

    f32x4 aReg[4], bReg[NB];

    // prefetch first K-tile
    #pragma unroll
    for (int i = 0; i < 4; ++i)
        aReg[i] = ANT ? __builtin_nontemporal_load((const f32x4*)aPtr[i])
                      : *(const f32x4*)aPtr[i];
    #pragma unroll
    for (int i = 0; i < NB; ++i) bReg[i] = *(const f32x4*)bPtr[i];

    for (int k0 = 0; k0 < K; k0 += KS) {
        // ---- convert prefetched tile -> LDS ----
        #pragma unroll
        for (int i = 0; i < 4; ++i) {
            half4 hv;
            hv[0] = (_Float16)(aReg[i][0] * ascale);
            hv[1] = (_Float16)(aReg[i][1] * ascale);
            hv[2] = (_Float16)(aReg[i][2] * ascale);
            hv[3] = (_Float16)(aReg[i][3] * ascale);
            *(half4*)&As[aLds[i]] = hv;
        }
        if (brow) {
            #pragma unroll
            for (int i = 0; i < NB; ++i) {
                Bs[bLds[i] +   0] = (_Float16)bReg[i][0];
                Bs[bLds[i] +  48] = (_Float16)bReg[i][1];
                Bs[bLds[i] +  96] = (_Float16)bReg[i][2];
                Bs[bLds[i] + 144] = (_Float16)bReg[i][3];
            }
        } else {
            #pragma unroll
            for (int i = 0; i < NB; ++i) {
                half4 hv;
                hv[0] = (_Float16)bReg[i][0]; hv[1] = (_Float16)bReg[i][1];
                hv[2] = (_Float16)bReg[i][2]; hv[3] = (_Float16)bReg[i][3];
                *(half4*)&Bs[bLds[i]] = hv;
            }
        }
        __syncthreads();

        // ---- prefetch next K-tile while WMMAs run ----
        if (k0 + KS < K) {
            #pragma unroll
            for (int i = 0; i < 4; ++i) {
                aPtr[i] += KS;
                aReg[i] = ANT ? __builtin_nontemporal_load((const f32x4*)aPtr[i])
                              : *(const f32x4*)aPtr[i];
            }
            #pragma unroll
            for (int i = 0; i < NB; ++i) {
                bPtr[i] += bStep;
                bReg[i] = *(const f32x4*)bPtr[i];
            }
        }

        // ---- fragments + WMMA ----
        const half8 alo = *(const half8*)&As[aFrag0];
        const half8 ahi = *(const half8*)&As[aFrag0 + 16];
        v16h a;
        #pragma unroll
        for (int i = 0; i < 8; ++i) { a[i] = alo[i]; a[8 + i] = ahi[i]; }

        #pragma unroll
        for (int nt = 0; nt < NTILE; ++nt) {
            const half8 blo = *(const half8*)&Bs[bFrag0 + nt * (16 * 48)];
            const half8 bhi = *(const half8*)&Bs[bFrag0 + nt * (16 * 48) + 16];
            v16h b;
            #pragma unroll
            for (int i = 0; i < 8; ++i) { b[i] = blo[i]; b[8 + i] = bhi[i]; }
            acc[nt] = __builtin_amdgcn_wmma_f32_16x16x32_f16(
                false, a, false, b, (short)0, acc[nt], false, false);
        }
        __syncthreads();
    }

    // ---- epilogue: scale, bias, activation, store ----
    #pragma unroll
    for (int nt = 0; nt < NTILE; ++nt) {
        const int n  = n0 + nt * 16 + lmod;
        const float bv = bias ? bias[n] : 0.0f;
        #pragma unroll
        for (int r = 0; r < 8; ++r) {
            const int m = m0 + wave * 16 + r + lhalf * 8;
            float v = acc[nt][r] * cscale + bv;
            if (act == 1)      v = v > 0.0f ? v : 0.0f;
            else if (act == 2) v = 1.0f / (1.0f + __expf(-v));
            float* dst = C + (size_t)m * ldc + n;
            if (CNT) __builtin_nontemporal_store(v, dst);
            else     *dst = v;
        }
    }
}

// z = mu + eps * exp(0.5 * log_var)
__global__ void reparam_kernel(const float* __restrict__ mu,
                               const float* __restrict__ lv,
                               const float* __restrict__ eps,
                               float* __restrict__ z, int n)
{
    const int i = blockIdx.x * blockDim.x + threadIdx.x;
    if (i < n) z[i] = mu[i] + eps[i] * __expf(0.5f * lv[i]);
}

extern "C" void kernel_launch(void* const* d_in, const int* in_sizes, int n_in,
                              void* d_out, int out_size, void* d_ws, size_t ws_size,
                              hipStream_t stream)
{
    (void)in_sizes; (void)n_in; (void)out_size; (void)ws_size;
    constexpr int Nn = 8192, F = 512, H = 128, L = 64;

    const float* x        = (const float*)d_in[0];
    const float* adj      = (const float*)d_in[1];
    const float* gc1_W    = (const float*)d_in[2];
    const float* gc1_b    = (const float*)d_in[3];
    const float* gc2_W    = (const float*)d_in[4];
    const float* gc2_b    = (const float*)d_in[5];
    const float* mu_W     = (const float*)d_in[6];
    const float* mu_b     = (const float*)d_in[7];
    const float* lv_W     = (const float*)d_in[8];
    const float* lv_b     = (const float*)d_in[9];
    const float* attr_W1  = (const float*)d_in[10];
    const float* attr_b1  = (const float*)d_in[11];
    const float* attr_W2  = (const float*)d_in[12];
    const float* attr_b2  = (const float*)d_in[13];
    const float* struct_W1= (const float*)d_in[14];
    const float* struct_b1= (const float*)d_in[15];
    const float* struct_W2= (const float*)d_in[16];
    const float* struct_b2= (const float*)d_in[17];
    const float* eps      = (const float*)d_in[18];

    float* out   = (float*)d_out;
    float* A_hat = out;                              // [N, N]
    float* x_hat = out + (size_t)Nn * Nn;            // [N, F]

    float* w  = (float*)d_ws;
    float* t1 = w;                    // [N, H]
    float* h1 = t1 + (size_t)Nn * H;
    float* t2 = h1 + (size_t)Nn * H;
    float* h2 = t2 + (size_t)Nn * H;
    float* mu = h2 + (size_t)Nn * H;  // [N, L]
    float* lv = mu + (size_t)Nn * L;
    float* z  = lv + (size_t)Nn * L;
    float* za = z  + (size_t)Nn * L;
    float* sa = za + (size_t)Nn * L;
    float* s  = sa + (size_t)Nn * L;

    auto gemm = [&](const float* A, int lda, const float* B, int bk, int bn,
                    const float* bias, float* C, int ldc, int K, int Ntot,
                    int act, float ascale, bool aNT, bool cNT) {
        const float cscale = 1.0f / ascale;
        if (Ntot % 128 == 0) {
            dim3 g(Nn / NROWS, Ntot / 128);
            if (aNT)
                gemm_wmma_f16<128, true, false><<<g, 256, 0, stream>>>(
                    A, lda, B, bk, bn, bias, C, ldc, K, act, ascale, cscale);
            else if (cNT)
                gemm_wmma_f16<128, false, true><<<g, 256, 0, stream>>>(
                    A, lda, B, bk, bn, bias, C, ldc, K, act, ascale, cscale);
            else
                gemm_wmma_f16<128, false, false><<<g, 256, 0, stream>>>(
                    A, lda, B, bk, bn, bias, C, ldc, K, act, ascale, cscale);
        } else {
            dim3 g(Nn / NROWS, Ntot / 64);
            gemm_wmma_f16<64, false, false><<<g, 256, 0, stream>>>(
                A, lda, B, bk, bn, bias, C, ldc, K, act, ascale, cscale);
        }
    };

    const float ADJ_SCALE = 16384.0f;  // keep adj (~1e-4) out of f16 denormals

    // Encoder
    gemm(x,   F,  gc1_W, H, 1, nullptr, t1, H, F,  H, 0, 1.0f,      false, false);
    gemm(adj, Nn, t1,    H, 1, gc1_b,   h1, H, Nn, H, 1, ADJ_SCALE, true,  false);
    gemm(h1,  H,  gc2_W, H, 1, nullptr, t2, H, H,  H, 0, 1.0f,      false, false);
    gemm(adj, Nn, t2,    H, 1, gc2_b,   h2, H, Nn, H, 1, ADJ_SCALE, true,  false);

    // Latent heads + reparameterization
    gemm(h2, H, mu_W, L, 1, mu_b, mu, L, H, L, 0, 1.0f, false, false);
    gemm(h2, H, lv_W, L, 1, lv_b, lv, L, H, L, 0, 1.0f, false, false);
    {
        const int n = Nn * L;
        reparam_kernel<<<(n + 255) / 256, 256, 0, stream>>>(mu, lv, eps, z, n);
    }

    // Attribute decoder -> x_hat (stream-out, NT stores)
    gemm(z,  L, attr_W1, L, 1, attr_b1, za,    L, L, L, 1, 1.0f, false, false);
    gemm(za, L, attr_W2, F, 1, attr_b2, x_hat, F, L, F, 0, 1.0f, false, true);

    // Structure decoder -> A_hat = sigmoid(s @ s^T) (stream-out, NT stores)
    gemm(z,  L, struct_W1, L, 1, struct_b1, sa, L, L, L, 1, 1.0f, false, false);
    gemm(sa, L, struct_W2, L, 1, struct_b2, s,  L, L, L, 0, 1.0f, false, false);
    gemm(s,  L, s, 1, L, nullptr, A_hat, Nn, L, Nn, 2, 1.0f, false, true);
}
